// IntraOptionPolicy_32693291057948
// MI455X (gfx1250) — matrix operand order repaired and checked by reading, not compiled, not run on hardware
//
#include <hip/hip_runtime.h>
#include <math.h>

typedef __attribute__((ext_vector_type(16))) __bf16 v16bf;
typedef __attribute__((ext_vector_type(8)))  float  v8f;

#define BB   256
#define UU   8
#define HID  128
#define OO   4
#define AA   7
#define VV   400
#define DK   64
#define DKC  32
#define EMBN 53824
#define CVN  6728

// ---------------------------------------------------------------------------
// Generic batched WMMA GEMM:  C[z] (+)= A[z] * B[z],  f32 in/out, bf16 compute
// One wave computes a 16x32 output strip (two 16x16 tiles sharing the A
// fragment -> 2 v_wmma per K-step, half the A traffic per FLOP).
// Main K-loop has zero guards: row/col indices are clamped in-range once
// (WMMA rows/cols are independent, so clamped-lane garbage only reaches
// store-masked elements). Single guarded K-tail iteration handles K%32.
// All control flow that skips WMMA is wave-uniform (EXEC all-ones at v_wmma).
// ---------------------------------------------------------------------------
template<bool ACC>
__global__ void gemm_wmma_bf16(const float* __restrict__ A,
                               const float* __restrict__ Bm,
                               float* __restrict__ C,
                               int M, int N, int Kt,
                               long aBatch, long bBatch, long cBatch,
                               int lda, int ldb, int ldc, int tilesN2)
{
    const int z = blockIdx.z;
    const float* __restrict__ Ab = A  + (long)z * aBatch;
    const float* __restrict__ Bb = Bm + (long)z * bBatch;
    float* __restrict__ Cb       = C  + (long)z * cBatch;

    const int wave = threadIdx.x >> 5;          // 4 waves per block
    const int lane = threadIdx.x & 31;
    const int tilesM = (M + 15) >> 4;
    const int tile = blockIdx.x * (blockDim.x >> 5) + wave;
    if (tile >= tilesM * tilesN2) return;       // wave-uniform
    const int tm = tile / tilesN2;
    const int tn = tile % tilesN2;              // 32-wide N strip index

    const int half = lane >> 4;                 // 0: lanes 0-15, 1: lanes 16-31
    const int l16  = lane & 15;

    int rowA = tm * 16 + l16;                   // A row owned by this lane
    if (rowA > M - 1) rowA = M - 1;             // clamp: OOB rows are store-masked
    int colB0 = tn * 32 + l16;                  // B col, first tile
    int colB1 = colB0 + 16;                     // B col, second tile
    const bool colOK0 = colB0 < N;
    const bool colOK1 = colB1 < N;
    if (!colOK0) colB0 = N - 1;                 // clamp: OOB cols are store-masked
    if (!colOK1) colB1 = N - 1;

    const float* __restrict__ Arow  = Ab + (long)rowA * lda;
    const float* __restrict__ Bcol0 = Bb + colB0;
    const float* __restrict__ Bcol1 = Bb + colB1;

    v8f acc0 = {}, acc1 = {};
    if (ACC) {
#pragma unroll
        for (int r = 0; r < 8; ++r) {
            int m  = tm * 16 + half * 8 + r;
            int mc = (m < M) ? m : (M - 1);
            float c0 = Cb[(long)mc * ldc + colB0];          // clamped, in-bounds
            float c1 = Cb[(long)mc * ldc + colB1];
            acc0[r] = ((m < M) && colOK0) ? c0 : 0.0f;      // data select
            acc1[r] = ((m < M) && colOK1) ? c1 : 0.0f;
        }
    }

    const int kmain = Kt & ~31;
    int k0 = 0;
    for (; k0 < kmain; k0 += 32) {
        if (k0 + 288 <= Kt)                                  // wave-uniform
            __builtin_prefetch(Arow + k0 + 256, 0, 1);
        // A: lane's 16 K-values = two contiguous 8-float runs -> 4x float4
        const float4* ap0 = (const float4*)(Arow + k0 + half * 8);
        const float4* ap1 = (const float4*)(Arow + k0 + 16 + half * 8);
        float4 a0 = ap0[0], a1 = ap0[1];
        float4 a2 = ap1[0], a3 = ap1[1];
        v16bf a, b0, b1;
        a[0]  = (__bf16)a0.x; a[1]  = (__bf16)a0.y; a[2]  = (__bf16)a0.z; a[3]  = (__bf16)a0.w;
        a[4]  = (__bf16)a1.x; a[5]  = (__bf16)a1.y; a[6]  = (__bf16)a1.z; a[7]  = (__bf16)a1.w;
        a[8]  = (__bf16)a2.x; a[9]  = (__bf16)a2.y; a[10] = (__bf16)a2.z; a[11] = (__bf16)a2.w;
        a[12] = (__bf16)a3.x; a[13] = (__bf16)a3.y; a[14] = (__bf16)a3.z; a[15] = (__bf16)a3.w;
        // B: 2x16 loads, each coalesced across the 16 lanes (consecutive cols)
#pragma unroll
        for (int e = 0; e < 16; ++e) {
            long ko = (long)(k0 + ((e < 8) ? 0 : 16) + (e & 7) + half * 8) * ldb;
            b0[e] = (__bf16)Bcol0[ko];
            b1[e] = (__bf16)Bcol1[ko];
        }
        acc0 = __builtin_amdgcn_wmma_f32_16x16x32_bf16(
                   false, a, false, b0, (short)0, acc0, false, false);
        acc1 = __builtin_amdgcn_wmma_f32_16x16x32_bf16(
                   false, a, false, b1, (short)0, acc1, false, false);
    }
    if (k0 < Kt) {                                           // K tail, <=1 iter
        v16bf a, b0, b1;
#pragma unroll
        for (int e = 0; e < 16; ++e) {
            int ka = k0 + ((e < 8) ? 0 : 16) + (e & 7) + half * 8;
            float av = Arow[(ka < Kt) ? ka : (Kt - 1)];      // clamped load
            a[e] = (__bf16)((ka < Kt) ? av : 0.0f);          // data select
            int kb = k0 + ((e < 8) ? 0 : 16) + (e & 7) + half * 8;
            long ko = (long)((kb < Kt) ? kb : (Kt - 1)) * ldb;
            float bv0 = Bcol0[ko];
            float bv1 = Bcol1[ko];
            b0[e] = (__bf16)((kb < Kt) ? bv0 : 0.0f);
            b1[e] = (__bf16)((kb < Kt) ? bv1 : 0.0f);
        }
        acc0 = __builtin_amdgcn_wmma_f32_16x16x32_bf16(
                   false, a, false, b0, (short)0, acc0, false, false);
        acc1 = __builtin_amdgcn_wmma_f32_16x16x32_bf16(
                   false, a, false, b1, (short)0, acc1, false, false);
    }

#pragma unroll
    for (int r = 0; r < 8; ++r) {
        int m = tm * 16 + half * 8 + r;
        if (m < M && colOK0) Cb[(long)m * ldc + colB0] = acc0[r];
        if (m < M && colOK1) Cb[(long)m * ldc + colB1] = acc1[r];
    }
}

// ---------------------------------------------------------------------------
// conv1 (3->16, 2x2, VALID) + ReLU + maxpool2   x: NHWC (B,64,64,3)
// out: (B,16,31,31)
// ---------------------------------------------------------------------------
__global__ void k_conv1_pool(const float* __restrict__ x,
                             const float* __restrict__ w,
                             const float* __restrict__ bias,
                             float* __restrict__ out)
{
    int idx = blockIdx.x * blockDim.x + threadIdx.x;
    const int total = BB * 16 * 31 * 31;
    if (idx >= total) return;
    int pw = idx % 31; int t = idx / 31;
    int ph = t % 31;   t /= 31;
    int oc = t % 16;   int b = t / 16;
    float best = -3.402823466e38f;
#pragma unroll
    for (int dy = 0; dy < 2; ++dy)
#pragma unroll
    for (int dx = 0; dx < 2; ++dx) {
        int oh = 2 * ph + dy, ow = 2 * pw + dx;
        float s = bias[oc];
#pragma unroll
        for (int ic = 0; ic < 3; ++ic)
#pragma unroll
        for (int kh = 0; kh < 2; ++kh)
#pragma unroll
        for (int kw = 0; kw < 2; ++kw)
            s += x[((b * 64 + oh + kh) * 64 + (ow + kw)) * 3 + ic]
               * w[((oc * 3 + ic) * 2 + kh) * 2 + kw];
        s = fmaxf(s, 0.0f);
        best = fmaxf(best, s);
    }
    out[idx] = best;
}

// conv2 (16->32, 2x2) + ReLU : in (B,16,31,31) -> out (B,32,30,30)
__global__ void k_conv2(const float* __restrict__ in,
                        const float* __restrict__ w,
                        const float* __restrict__ bias,
                        float* __restrict__ out)
{
    int idx = blockIdx.x * blockDim.x + threadIdx.x;
    const int total = BB * 32 * 30 * 30;
    if (idx >= total) return;
    int ow = idx % 30; int t = idx / 30;
    int oh = t % 30;   t /= 30;
    int oc = t % 32;   int b = t / 32;
    float s = bias[oc];
    for (int ic = 0; ic < 16; ++ic)
#pragma unroll
        for (int kh = 0; kh < 2; ++kh)
#pragma unroll
        for (int kw = 0; kw < 2; ++kw)
            s += in[((b * 16 + ic) * 31 + oh + kh) * 31 + (ow + kw)]
               * w[((oc * 16 + ic) * 2 + kh) * 2 + kw];
    out[idx] = fmaxf(s, 0.0f);
}

// conv3 (32->64, 2x2) + ReLU : in (B,32,30,30) -> emb (B, 64*29*29)
__global__ void k_conv3(const float* __restrict__ in,
                        const float* __restrict__ w,
                        const float* __restrict__ bias,
                        float* __restrict__ emb)
{
    int idx = blockIdx.x * blockDim.x + threadIdx.x;
    const int total = BB * 64 * 29 * 29;
    if (idx >= total) return;
    int ow = idx % 29; int t = idx / 29;
    int oh = t % 29;   t /= 29;
    int oc = t % 64;   int b = t / 64;
    float s = bias[oc];
    for (int ic = 0; ic < 32; ++ic)
#pragma unroll
        for (int kh = 0; kh < 2; ++kh)
#pragma unroll
        for (int kw = 0; kw < 2; ++kw)
            s += in[((b * 32 + ic) * 30 + oh + kh) * 30 + (ow + kw)]
               * w[((oc * 32 + ic) * 2 + kh) * 2 + kw];
    emb[(long)b * EMBN + oc * 841 + oh * 29 + ow] = fmaxf(s, 0.0f);
}

// q[b,u,k] = sum_h hx[b,u,h] * sum_o option[b,o]*Wq[o,u,h,k]
__global__ void k_q(const float* __restrict__ hx, const float* __restrict__ option,
                    const float* __restrict__ Wq, float* __restrict__ q)
{
    int idx = blockIdx.x * blockDim.x + threadIdx.x;
    if (idx >= BB * UU * DK) return;
    int k = idx & 63; int u = (idx >> 6) & 7; int b = idx >> 9;
    float s = 0.0f;
    for (int h = 0; h < HID; ++h) {
        float wq = 0.0f;
#pragma unroll
        for (int o = 0; o < OO; ++o)
            wq += option[b * OO + o] * Wq[(((long)o * UU + u) * HID + h) * DK + k];
        s += hx[(b * UU + u) * HID + h] * wq;
    }
    q[idx] = s;
}

// softmax over n=2 scores; k_in[:,1]=bk, v_in[:,1]=bv since xin[:,1]==0
__global__ void k_probs(const float* __restrict__ q, const float* __restrict__ kin0,
                        const float* __restrict__ bk, float* __restrict__ p01)
{
    int idx = blockIdx.x * blockDim.x + threadIdx.x;
    if (idx >= BB * UU) return;
    int b = idx >> 3;
    float s0 = 0.0f, s1 = 0.0f;
    for (int k = 0; k < DK; ++k) {
        float qq = q[idx * DK + k];
        s0 += qq * (kin0[b * DK + k] + bk[k]);
        s1 += qq * bk[k];
    }
    s0 *= 0.125f; s1 *= 0.125f;                 // 1/sqrt(64)
    float mx = fmaxf(s0, s1);
    float e0 = expf(s0 - mx), e1 = expf(s1 - mx);
    float inv = 1.0f / (e0 + e1);
    p01[idx * 2 + 0] = e0 * inv;
    p01[idx * 2 + 1] = e1 * inv;
}

// attn_in[b,u,v] = p0*(v_in0[b,v]+bv[v]) + p1*bv[v]
__global__ void k_attn(const float* __restrict__ p01, const float* __restrict__ vin0,
                       const float* __restrict__ bv, float* __restrict__ attn)
{
    int idx = blockIdx.x * blockDim.x + threadIdx.x;
    if (idx >= BB * UU * VV) return;
    int v = idx % VV; int bu = idx / VV; int b = bu >> 3;
    float p0 = p01[bu * 2 + 0], p1 = p01[bu * 2 + 1];
    attn[idx] = p0 * (vin0[b * VV + v] + bv[v]) + p1 * bv[v];
}

// top-4-of-8 mask with lax.top_k tie-breaking (earlier index wins on ties)
__global__ void k_mask(const float* __restrict__ p01, float* __restrict__ mask)
{
    int b = blockIdx.x * blockDim.x + threadIdx.x;
    if (b >= BB) return;
    for (int u = 0; u < UU; ++u) {
        float r = p01[(b * UU + u) * 2];
        int rank = 0;
        for (int j = 0; j < UU; ++j) {
            float rj = p01[(b * UU + j) * 2];
            rank += (rj > r) || (rj == r && j < u);
        }
        mask[b * UU + u] = (rank < 4) ? 1.0f : 0.0f;
    }
}

__device__ __forceinline__ float sigf(float x) { return 1.0f / (1.0f + expf(-x)); }

// LSTM cell elementwise
__global__ void k_lstm(const float* __restrict__ pre, const float* __restrict__ b_lstm,
                       const float* __restrict__ cx, float* __restrict__ cnew,
                       float* __restrict__ hnew)
{
    int idx = blockIdx.x * blockDim.x + threadIdx.x;
    if (idx >= BB * UU * HID) return;
    int h = idx & 127; int u = (idx >> 7) & 7;
    int base = (idx >> 7) * (4 * HID);
    int bb = u * (4 * HID);
    float ig = pre[base + h]            + b_lstm[bb + h];
    float fg = pre[base + HID + h]      + b_lstm[bb + HID + h];
    float gg = pre[base + 2 * HID + h]  + b_lstm[bb + 2 * HID + h];
    float og = pre[base + 3 * HID + h]  + b_lstm[bb + 3 * HID + h];
    float c  = sigf(fg) * cx[idx] + sigf(ig) * tanhf(gg);
    cnew[idx] = c;
    hnew[idx] = sigf(og) * tanhf(c);
}

// comm scores softmax(U=8) * mask
__global__ void k_cp(const float* __restrict__ qc, const float* __restrict__ kc,
                     const float* __restrict__ mask, float* __restrict__ cp)
{
    int idx = blockIdx.x * blockDim.x + threadIdx.x;
    if (idx >= BB * UU) return;
    int b = idx >> 3;
    float cs[UU]; float mx = -3.402823466e38f;
    for (int n = 0; n < UU; ++n) {
        float s = 0.0f;
        for (int k = 0; k < DKC; ++k)
            s += qc[idx * DKC + k] * kc[(b * UU + n) * DKC + k];
        s *= 0.17677669529663687f;              // 1/sqrt(32)
        cs[n] = s; mx = fmaxf(mx, s);
    }
    float sum = 0.0f;
    for (int n = 0; n < UU; ++n) { cs[n] = expf(cs[n] - mx); sum += cs[n]; }
    float m = mask[idx] / sum;
    for (int n = 0; n < UU; ++n) cp[idx * UU + n] = cs[n] * m;
}

// ctx[b,u,v] = sum_n cp[b,u,n]*vc[b,n,v]
__global__ void k_ctx(const float* __restrict__ cp, const float* __restrict__ vc,
                      float* __restrict__ ctx)
{
    long idx = (long)blockIdx.x * blockDim.x + threadIdx.x;
    if (idx >= (long)BB * UU * CVN) return;
    int v = (int)(idx % CVN); int bu = (int)(idx / CVN); int b = bu >> 3;
    float s = 0.0f;
#pragma unroll
    for (int n = 0; n < UU; ++n)
        s += cp[bu * UU + n] * vc[((long)(b * UU + n)) * CVN + v];
    ctx[idx] = s;
}

// gate hy/cy and write to output
__global__ void k_final(const float* __restrict__ hnew, const float* __restrict__ dctx,
                        const float* __restrict__ cnew, const float* __restrict__ hx,
                        const float* __restrict__ cx, const float* __restrict__ mask,
                        float* __restrict__ out)
{
    int idx = blockIdx.x * blockDim.x + threadIdx.x;
    if (idx >= BB * UU * HID) return;
    float m = mask[idx >> 7];
    float hc = hnew[idx] + dctx[idx];
    float hy = m * hc + (1.0f - m) * hx[idx];
    float cy = m * cnew[idx] + (1.0f - m) * cx[idx];
    out[BB * OO * AA + idx] = hy;
    out[BB * OO * AA + BB * UU * HID + idx] = cy;
}

// hsum[b,h] = sum_u hy[b,u,h]
__global__ void k_hsum(const float* __restrict__ out, float* __restrict__ hsum)
{
    int idx = blockIdx.x * blockDim.x + threadIdx.x;
    if (idx >= BB * HID) return;
    int h = idx & 127; int b = idx >> 7;
    float s = 0.0f;
#pragma unroll
    for (int u = 0; u < UU; ++u)
        s += out[BB * OO * AA + (b * UU + u) * HID + h];
    hsum[idx] = s;
}

// policy logits + softmax over A=7
__global__ void k_policy(const float* __restrict__ hsum, const float* __restrict__ pi_w,
                         const float* __restrict__ pi_b, float* __restrict__ out)
{
    int idx = blockIdx.x * blockDim.x + threadIdx.x;
    if (idx >= BB * OO) return;
    int o = idx & 3; int b = idx >> 2;
    float lg[AA]; float mx = -3.402823466e38f;
    for (int a = 0; a < AA; ++a) {
        float s = pi_b[o * AA + a];
        for (int h = 0; h < HID; ++h)
            s += hsum[b * HID + h] * pi_w[(o * HID + h) * AA + a];
        lg[a] = s; mx = fmaxf(mx, s);
    }
    float sum = 0.0f;
    for (int a = 0; a < AA; ++a) { lg[a] = expf(lg[a] - mx); sum += lg[a]; }
    float inv = 1.0f / sum;
    for (int a = 0; a < AA; ++a)
        out[(b * OO + o) * AA + a] = lg[a] * inv;
}

// ---------------------------------------------------------------------------
extern "C" void kernel_launch(void* const* d_in, const int* in_sizes, int n_in,
                              void* d_out, int out_size, void* d_ws, size_t ws_size,
                              hipStream_t stream)
{
    (void)in_sizes; (void)n_in; (void)out_size; (void)ws_size;
    const float* x       = (const float*)d_in[0];
    const float* hx      = (const float*)d_in[1];
    const float* cx      = (const float*)d_in[2];
    const float* option  = (const float*)d_in[3];
    const float* c1w     = (const float*)d_in[4];
    const float* c1b     = (const float*)d_in[5];
    const float* c2w     = (const float*)d_in[6];
    const float* c2b     = (const float*)d_in[7];
    const float* c3w     = (const float*)d_in[8];
    const float* c3b     = (const float*)d_in[9];
    const float* Wk      = (const float*)d_in[10];
    const float* bk      = (const float*)d_in[11];
    const float* Wv      = (const float*)d_in[12];
    const float* bv      = (const float*)d_in[13];
    const float* Wq      = (const float*)d_in[14];
    const float* Wi      = (const float*)d_in[15];
    const float* Wh      = (const float*)d_in[16];
    const float* b_lstm  = (const float*)d_in[17];
    const float* Wqc     = (const float*)d_in[18];
    const float* Wkc     = (const float*)d_in[19];
    const float* Wvc     = (const float*)d_in[20];
    const float* Wout    = (const float*)d_in[21];
    const float* pi_w    = (const float*)d_in[22];
    const float* pi_b    = (const float*)d_in[23];
    float* out = (float*)d_out;
    float* ws  = (float*)d_ws;

    // workspace layout (floats); emb region reused as vc (identical size),
    // conv buffer region reused as ctx.
    size_t off = 0;
    float* emb_vc = ws + off; off += 13778944;          // emb (B,EMB) / vc (B,U,CV)
    float* big2   = ws + off; off += 13778944;          // h1+h2 / ctx
    float* h1 = big2;                                   // (B,16,31,31) = 3,936,256
    float* h2 = big2 + 3936256;                         // (B,32,30,30) = 7,372,800
    float* ctx = big2;                                  // (B,U,CV)
    float* kin0 = ws + off; off += 16384;
    float* vin0 = ws + off; off += 102400;
    float* qbuf = ws + off; off += 131072;
    float* p01  = ws + off; off += 4096;
    float* attn = ws + off; off += 819200;
    float* maskb= ws + off; off += 2048;
    float* pre  = ws + off; off += 1048576;
    float* cnew = ws + off; off += 262144;
    float* hnew = ws + off; off += 262144;
    float* qc   = ws + off; off += 65536;
    float* kc   = ws + off; off += 65536;
    float* cpb  = ws + off; off += 16384;
    float* dctx = ws + off; off += 262144;
    float* hsum = ws + off; off += 32768;

    auto gemm = [&](const float* A, const float* Bm, float* C, int M, int N, int Kt,
                    long aB, long bB, long cB, int lda, int ldb, int ldc,
                    int batch, bool acc) {
        int tilesM = (M + 15) / 16, tilesN2 = (N + 31) / 32;
        int tiles = tilesM * tilesN2;
        dim3 grid((tiles + 3) / 4, 1, batch);
        if (acc)
            gemm_wmma_bf16<true ><<<grid, 128, 0, stream>>>(A, Bm, C, M, N, Kt,
                aB, bB, cB, lda, ldb, ldc, tilesN2);
        else
            gemm_wmma_bf16<false><<<grid, 128, 0, stream>>>(A, Bm, C, M, N, Kt,
                aB, bB, cB, lda, ldb, ldc, tilesN2);
    };

    // conv stack
    k_conv1_pool<<<(BB*16*31*31 + 255)/256, 256, 0, stream>>>(x, c1w, c1b, h1);
    k_conv2     <<<(BB*32*30*30 + 255)/256, 256, 0, stream>>>(h1, c2w, c2b, h2);
    k_conv3     <<<(BB*64*29*29 + 255)/256, 256, 0, stream>>>(h2, c3w, c3b, emb_vc);

    // k_in / v_in  (row n=0 only; n=1 handled analytically with biases)
    gemm(emb_vc, Wk, kin0, BB, DK, EMBN, 0, 0, 0, EMBN, DK, DK, 1, false);
    gemm(emb_vc, Wv, vin0, BB, VV, EMBN, 0, 0, 0, EMBN, VV, VV, 1, false);

    // option-conditioned query, attention over n=2, mask
    k_q    <<<(BB*UU*DK + 255)/256, 256, 0, stream>>>(hx, option, Wq, qbuf);
    k_probs<<<(BB*UU + 255)/256, 256, 0, stream>>>(qbuf, kin0, bk, p01);
    k_attn <<<(BB*UU*VV + 255)/256, 256, 0, stream>>>(p01, vin0, bv, attn);
    k_mask <<<(BB + 255)/256, 256, 0, stream>>>(p01, maskb);

    // pre = attn @ Wi  (batched over u), then += hx @ Wh
    gemm(attn, Wi, pre, BB, 4*HID, VV, 400, 204800, 512, UU*VV, 4*HID, UU*4*HID, UU, false);
    gemm(hx,   Wh, pre, BB, 4*HID, HID, 128, 65536, 512, UU*HID, 4*HID, UU*4*HID, UU, true);

    // LSTM
    k_lstm<<<(BB*UU*HID + 255)/256, 256, 0, stream>>>(pre, b_lstm, cx, cnew, hnew);

    // comm attention projections (batched over u)
    gemm(hnew, Wqc, qc, BB, DKC, HID, 128, 4096,   32,   UU*HID, DKC, UU*DKC, UU, false);
    gemm(hnew, Wkc, kc, BB, DKC, HID, 128, 4096,   32,   UU*HID, DKC, UU*DKC, UU, false);
    gemm(hnew, Wvc, emb_vc, BB, CVN, HID, 128, (long)HID*CVN, CVN, UU*HID, CVN, UU*CVN, UU, false);

    k_cp <<<(BB*UU + 255)/256, 256, 0, stream>>>(qc, kc, maskb, cpb);
    k_ctx<<<(int)(((long)BB*UU*CVN + 255)/256), 256, 0, stream>>>(cpb, emb_vc, ctx);

    // dctx = ctx @ Wout (batched over u)
    gemm(ctx, Wout, dctx, BB, HID, CVN, CVN, (long)CVN*HID, HID, UU*CVN, HID, UU*HID, UU, false);

    // gating, output hy/cy, policy head
    k_final <<<(BB*UU*HID + 255)/256, 256, 0, stream>>>(hnew, dctx, cnew, hx, cx, maskb, out);
    k_hsum  <<<(BB*HID + 255)/256, 256, 0, stream>>>(out, hsum);
    k_policy<<<(BB*OO + 255)/256, 256, 0, stream>>>(hsum, pi_w, pi_b, out);
}
